// Rnn_91122026151947
// MI455X (gfx1250) — matrix operand order repaired and checked by reading, not compiled, not run on hardware
//
#include <hip/hip_runtime.h>

#define BATCH 64
#define SEQ   512
#define INP   1024
#define HID   1024
#define GATES 4096   // 4*HID, torch gate order i,f,g,o

#define NWG    64    // one WG per 16 hidden units
#define NTHR   256   // 8 wave32
#define KC     128   // K-chunk staged in LDS
#define NCHUNK 16    // 2048 total K (h:1024 then x:1024) / 128

typedef __attribute__((ext_vector_type(16))) __bf16 v16bf;
typedef __attribute__((ext_vector_type(8)))  float  v8f;
typedef __attribute__((ext_vector_type(4)))  unsigned v4u;
typedef __attribute__((ext_vector_type(8)))  int      v8i;
typedef __attribute__((ext_vector_type(4)))  int      v4i;

#if __has_builtin(__builtin_amdgcn_tensor_load_to_lds) && \
    __has_builtin(__builtin_amdgcn_s_wait_tensorcnt)
#define HAVE_TDM 1
#else
#define HAVE_TDM 0
#endif

union Frag { v16bf v; unsigned u[8]; };
union AccU { v8f  v; float    f[8]; };

__device__ __forceinline__ unsigned short f2bf(float f) {
  unsigned x = __builtin_bit_cast(unsigned, f);
  x += 0x7FFFu + ((x >> 16) & 1u);      // round-to-nearest-even
  return (unsigned short)(x >> 16);
}

__device__ __forceinline__ float sigmoidf_fast(float x) {
  return 1.0f / (1.0f + __expf(-x));
}

// Per-lane async DMA: 16B global -> LDS, tracked by ASYNCcnt (no VGPR data).
__device__ __forceinline__ void async_copy_b128(unsigned lds_off,
                                                unsigned long long gaddr) {
  asm volatile("global_load_async_to_lds_b128 %0, %1, off"
               :: "v"(lds_off), "v"(gaddr)
               : "memory");
}

__device__ __forceinline__ void wait_asynccnt0() {
  asm volatile("s_wait_asynccnt 0x0" ::: "memory");
}

// Device-wide sense-reversing barrier (persistent kernel).
__device__ __forceinline__ void grid_sync(unsigned* cnt, unsigned* gen, unsigned nwg) {
  __builtin_amdgcn_fence(__ATOMIC_RELEASE, "agent");   // flush our h/c stores
  __syncthreads();
  if (threadIdx.x == 0) {
    unsigned g = __hip_atomic_load(gen, __ATOMIC_RELAXED, __HIP_MEMORY_SCOPE_AGENT);
    unsigned a = __hip_atomic_fetch_add(cnt, 1u, __ATOMIC_ACQ_REL, __HIP_MEMORY_SCOPE_AGENT);
    if (a == nwg - 1u) {
      __hip_atomic_store(cnt, 0u, __ATOMIC_RELAXED, __HIP_MEMORY_SCOPE_AGENT);
      __hip_atomic_fetch_add(gen, 1u, __ATOMIC_RELEASE, __HIP_MEMORY_SCOPE_AGENT);
    } else {
      while (__hip_atomic_load(gen, __ATOMIC_RELAXED, __HIP_MEMORY_SCOPE_AGENT) == g) {
        __builtin_amdgcn_s_sleep(2);
      }
    }
  }
  __syncthreads();
  __builtin_amdgcn_fence(__ATOMIC_ACQUIRE, "agent");   // invalidate stale h lines
}

// -------- one-time prep: fp32 -> bf16 weights/x, fused bias, zero state -----
__global__ void prep_kernel(const float* __restrict__ x,
                            const float* __restrict__ Wih,
                            const float* __restrict__ Whh,
                            const float* __restrict__ bih,
                            const float* __restrict__ bhh,
                            unsigned short* __restrict__ xbf,
                            unsigned short* __restrict__ wihbf,
                            unsigned short* __restrict__ whhbf,
                            float* __restrict__ bias,
                            float* __restrict__ cbuf,
                            unsigned short* __restrict__ h0,
                            unsigned short* __restrict__ h1,
                            unsigned* __restrict__ syncv) {
  const size_t XN = (size_t)BATCH * SEQ * INP;
  const size_t WN = (size_t)GATES * INP;
  const size_t CN = (size_t)BATCH * HID;
  const size_t total = XN + 2 * WN + GATES + 3 * CN + 8;
  size_t stride = (size_t)gridDim.x * blockDim.x;
  for (size_t i = (size_t)blockIdx.x * blockDim.x + threadIdx.x; i < total; i += stride) {
    size_t k = i;
    if (k < XN) { xbf[k] = f2bf(x[k]); continue; }
    k -= XN;
    if (k < WN) { wihbf[k] = f2bf(Wih[k]); continue; }
    k -= WN;
    if (k < WN) { whhbf[k] = f2bf(Whh[k]); continue; }
    k -= WN;
    if (k < GATES) { bias[k] = bih[k] + bhh[k]; continue; }
    k -= GATES;
    if (k < CN) { cbuf[k] = 0.0f; continue; }
    k -= CN;
    if (k < CN) { h0[k] = 0; continue; }
    k -= CN;
    if (k < CN) { h1[k] = 0; continue; }
    k -= CN;
    syncv[k] = 0u;
  }
}

// -------- persistent fused LSTM -----------------------------------------
// WG w owns hidden units [w*16, w*16+16) -> 64 gate rows {g*1024 + w*16 + r}.
// Per step: gates[64B x 64N] = [h | x_t] (64x2048) @ [Whh | Wih]^T slab, WMMA bf16.
// A chunk staged via global_load_async_to_lds_b128 (ASYNCcnt);
// B (weight slab) staged via TDM tensor_load_to_lds 3D tile (TENSORcnt).
__global__ void __launch_bounds__(NTHR, 1)
lstm_kernel(const unsigned short* __restrict__ xbf,
            const unsigned short* __restrict__ wihbf,
            const unsigned short* __restrict__ whhbf,
            const float* __restrict__ bias,
            float* __restrict__ cbuf,
            unsigned short* __restrict__ h0,
            unsigned short* __restrict__ h1,
            unsigned* __restrict__ syncv,
            float* __restrict__ out) {
  __shared__ unsigned short A_lds[64 * KC];       // 16 KB: [h|x] rows (batch)
  __shared__ unsigned short B_lds[64 * KC];       // 16 KB: W slab rows (4 gates x 16)
  __shared__ float gates_lds[4 * 64 * 16];        // 16 KB: post-GEMM gates

  const int tid  = threadIdx.x;
  const int wg   = blockIdx.x;          // 0..63
  const int hid0 = wg * 16;
  const int lane = tid & 31;
  const int wv   = tid >> 5;            // 0..7
  const int m_tile = wv & 3;            // batch tile (16 rows each)
  const int g0 = (wv >> 2) * 2;         // gates {0,1} or {2,3}
  const int g1 = g0 + 1;
  const int nlane = lane & 15;
  const int khalf = lane >> 4;

  const unsigned ldsA0 = (unsigned)(uintptr_t)(void*)&A_lds[0];
  const unsigned ldsB0 = (unsigned)(uintptr_t)(void*)&B_lds[0];

  // A-fragment K offsets per ISA 16-bit A 16x32 layout
  const int kofsA[8] = {0, 2, 4, 6, 16, 18, 20, 22};

  int pp = 0;
  for (int t = 0; t < SEQ; ++t) {
    const unsigned short* hsrc = pp ? h1 : h0;
    unsigned short*       hdst = pp ? h0 : h1;

    AccU acc0, acc1;
#pragma unroll
    for (int r = 0; r < 8; ++r) { acc0.f[r] = 0.0f; acc1.f[r] = 0.0f; }

    for (int kc = 0; kc < NCHUNK; ++kc) {
      __syncthreads();                       // prior chunk fully consumed
      const bool hphase = (kc < 8);
      const int  k0 = (hphase ? kc : kc - 8) * KC;

      // ---- A chunk: 1024 x 16B lane-level async DMA global -> LDS ----
#pragma unroll
      for (int j = 0; j < 4; ++j) {
        const int slot = j * 256 + tid;       // 0..1023
        const int row  = slot >> 4;           // batch row 0..63
        const int c16  = slot & 15;           // 16B column within row
        const unsigned short* srcA = hphase
            ? (hsrc + (size_t)row * HID + k0 + c16 * 8)
            : (xbf + (size_t)row * ((size_t)SEQ * INP) + (size_t)t * INP + k0 + c16 * 8);
        async_copy_b128(ldsA0 + (unsigned)slot * 16u,
                        (unsigned long long)(uintptr_t)srcA);
        if (j == 0) __builtin_prefetch(srcA + KC, 0, 3);
      }

      // ---- B chunk: TDM 3D tile (x=128 K, y=16 rows, z=4 gate blocks) ----
      {
        const unsigned short* Wsel = hphase ? whhbf : wihbf;
        const unsigned long long ga = (unsigned long long)(uintptr_t)
            (Wsel + (size_t)hid0 * 1024 + k0);   // tile start (bytes implicit *2)
#if HAVE_TDM
        if (wv == 0) {
          v4u g0v;
          g0v[0] = 1u;                                   // count=1, user D#
          g0v[1] = ldsB0;                                // lds_addr (bytes)
          g0v[2] = (unsigned)(ga & 0xFFFFFFFFu);         // global_addr[31:0]
          g0v[3] = (unsigned)((ga >> 32) & 0x01FFFFFFu)  // global_addr[56:32]
                   | (2u << 30);                         // type=2 (image)
          v8i g1v;
          g1v[0] = (int)(1u << 16);                      // data_size=1 -> 2B elems
          g1v[1] = (int)((1024u & 0xFFFFu) << 16);       // tensor_dim0[15:0]
          g1v[2] = (int)((1024u >> 16) | ((4096u & 0xFFFFu) << 16)); // td0 hi | td1 lo
          g1v[3] = (int)((4096u >> 16) | (128u << 16));  // td1 hi | tile_dim0=128
          g1v[4] = (int)(16u | (4u << 16));              // tile_dim1=16 | tile_dim2=4
          g1v[5] = (int)1024;                            // tensor_dim0_stride lo32
          g1v[6] = (int)((1048576u & 0xFFFFu) << 16);    // s0 hi16=0 | s1[15:0] (1M elems)
          g1v[7] = (int)(1048576u >> 16);                // s1[47:16]
          v4i g2v;
          g2v[0] = 4;                                    // tensor_dim2 = 4 gates
          g2v[1] = 0;                                    // tensor_dim3
          g2v[2] = 0;                                    // tensor_dim2_stride lo32
          g2v[3] = 0;                                    // s2 hi | tile_dim3=0
          v4i g3v = {0, 0, 0, 0};
          v8i g4v = {0, 0, 0, 0, 0, 0, 0, 0};            // unused 5th group (6-arg form)
          __builtin_amdgcn_tensor_load_to_lds(g0v, g1v, g2v, g3v, g4v, 0);
          __builtin_amdgcn_s_wait_tensorcnt(0);
        }
#else
        // Fallback: lane-level async DMA for the weight slab.
#pragma unroll
        for (int j = 0; j < 4; ++j) {
          const int slot = j * 256 + tid;
          const int row  = slot >> 4;           // g*16 + r
          const int c16  = slot & 15;
          const int gg = row >> 4, rr = row & 15;
          const unsigned short* srcB =
              Wsel + (size_t)(gg * HID + hid0 + rr) * 1024 + k0 + c16 * 8;
          async_copy_b128(ldsB0 + (unsigned)slot * 16u,
                          (unsigned long long)(uintptr_t)srcB);
        }
        (void)ga;
#endif
      }

      wait_asynccnt0();                 // our lane DMAs landed in LDS
      __syncthreads();                  // everyone's (and TDM's) data visible

      // ---- 4 K-steps of 32, 2 WMMA each (gate pair) ----
#pragma unroll
      for (int ks = 0; ks < 4; ++ks) {
        const int kl = ks * 32;
        Frag a, b0, b1;
#pragma unroll
        for (int v = 0; v < 8; ++v) {
          a.u[v]  = *(const unsigned*)&A_lds[(m_tile * 16 + nlane) * KC + kl +
                                             kofsA[v] + 8 * khalf];
          const int kb = kl + kofsA[v] + 8 * khalf;
          b0.u[v] = *(const unsigned*)&B_lds[(g0 * 16 + nlane) * KC + kb];
          b1.u[v] = *(const unsigned*)&B_lds[(g1 * 16 + nlane) * KC + kb];
        }
        acc0.v = __builtin_amdgcn_wmma_f32_16x16x32_bf16(
            false, a.v, false, b0.v, (short)0, acc0.v, false, false);
        acc1.v = __builtin_amdgcn_wmma_f32_16x16x32_bf16(
            false, a.v, false, b1.v, (short)0, acc1.v, false, false);
      }
    }
    __syncthreads();
    // ---- spill gate tiles (+bias) to LDS; C/D layout: M = r + 8*khalf ----
#pragma unroll
    for (int r = 0; r < 8; ++r) {
      const int m = m_tile * 16 + r + 8 * khalf;
      gates_lds[g0 * 1024 + m * 16 + nlane] = acc0.f[r] + bias[g0 * HID + hid0 + nlane];
      gates_lds[g1 * 1024 + m * 16 + nlane] = acc1.f[r] + bias[g1 * HID + hid0 + nlane];
    }
    __syncthreads();
    // ---- fused LSTM cell update for 64 batch x 16 hidden ----
#pragma unroll
    for (int j = 0; j < 4; ++j) {
      const int idx = j * 256 + tid;     // 0..1023
      const int m = idx >> 4;
      const int n = idx & 15;
      const float iv = gates_lds[idx];
      const float fv = gates_lds[1024 + idx];
      const float gv = gates_lds[2048 + idx];
      const float ov = gates_lds[3072 + idx];
      const size_t ci = (size_t)m * HID + hid0 + n;
      const float c_old = cbuf[ci];
      const float cn = sigmoidf_fast(fv) * c_old + sigmoidf_fast(iv) * tanhf(gv);
      const float hn = sigmoidf_fast(ov) * tanhf(cn);
      cbuf[ci] = cn;
      out[(size_t)m * SEQ * HID + (size_t)t * HID + hid0 + n] = hn;
      hdst[ci] = f2bf(hn);
    }
    grid_sync(syncv, syncv + 1, NWG);
    pp ^= 1;
  }
}

extern "C" void kernel_launch(void* const* d_in, const int* in_sizes, int n_in,
                              void* d_out, int out_size, void* d_ws, size_t ws_size,
                              hipStream_t stream) {
  const float* x   = (const float*)d_in[0];
  const float* Wih = (const float*)d_in[1];
  const float* Whh = (const float*)d_in[2];
  const float* bih = (const float*)d_in[3];
  const float* bhh = (const float*)d_in[4];
  float* out = (float*)d_out;

  char* ws = (char*)d_ws;
  size_t off = 0;
  auto carve = [&](size_t bytes) -> char* {
    char* p = ws + off;
    off += (bytes + 255) & ~(size_t)255;
    return p;
  };
  unsigned short* xbf   = (unsigned short*)carve((size_t)BATCH * SEQ * INP * 2);
  unsigned short* wihbf = (unsigned short*)carve((size_t)GATES * INP * 2);
  unsigned short* whhbf = (unsigned short*)carve((size_t)GATES * HID * 2);
  float*          bias  = (float*)carve((size_t)GATES * 4);
  float*          cbuf  = (float*)carve((size_t)BATCH * HID * 4);
  unsigned short* h0    = (unsigned short*)carve((size_t)BATCH * HID * 2);
  unsigned short* h1    = (unsigned short*)carve((size_t)BATCH * HID * 2);
  unsigned*       syncv = (unsigned*)carve(8 * sizeof(unsigned));

  prep_kernel<<<2048, 256, 0, stream>>>(x, Wih, Whh, bih, bhh,
                                        xbf, wihbf, whhbf, bias, cbuf, h0, h1, syncv);
  lstm_kernel<<<NWG, NTHR, 0, stream>>>(xbf, wihbf, whhbf, bias, cbuf,
                                        h0, h1, syncv, out);
}
